// Campbell_47261820125454
// MI455X (gfx1250) — compile-verified, matrix operand order checked by reading, not run on hardware
//
#include <hip/hip_runtime.h>
#include <stdint.h>

// ---------------------------------------------------------------------------
// Types for CDNA5 WMMA (wave32): A/B = 16 x bf16 (8 VGPRs), C/D = 8 x f32.
// ---------------------------------------------------------------------------
typedef __bf16 v16bf __attribute__((ext_vector_type(16)));
typedef float  v8f   __attribute__((ext_vector_type(8)));

union FragU {
    uint4          q[2];   // 32 bytes
    v16bf          v;
    unsigned short u[16];
};

__device__ __forceinline__ unsigned short f2bf(float f) {
    unsigned int u = __builtin_bit_cast(unsigned int, f);
    unsigned int r = (u + 0x7FFFu + ((u >> 16) & 1u)) >> 16;   // RNE
    return (unsigned short)r;
}

// Fragment load straight from global memory (used by attention).
//   lane<16 : row = row0+lane,    K chunks {0..7, 16..23}
//   lane>=16: row = row0+lane-16, K chunks {8..15, 24..31}
__device__ __forceinline__ v16bf load_frag(const unsigned short* base,
                                           int row0, int k0, int ldk) {
    int lane = threadIdx.x & 31;
    int r    = row0 + (lane & 15);
    int kb   = k0 + ((lane >> 4) << 3);
    const unsigned short* p = base + (size_t)r * ldk + kb;
    FragU f;
    f.q[0] = *(const uint4*)(p);
    f.q[1] = *(const uint4*)(p + 16);
    return f.v;
}

__device__ __forceinline__ v8f wmma_bf16(v16bf a, v16bf b, v8f c) {
    return __builtin_amdgcn_wmma_f32_16x16x32_bf16(false, a, false, b,
                                                   (short)0, c, false, false);
}

// Async global -> LDS copy, 16 bytes per lane (GV mode), tracked by ASYNCcnt.
__device__ __forceinline__ void async_b128(unsigned int lds_byte, const void* g) {
    asm volatile("global_load_async_to_lds_b128 %0, %1, off"
                 :: "v"(lds_byte),
                    "v"((unsigned long long)(uintptr_t)g)
                 : "memory");
}

// ---------------------------------------------------------------------------
// f32 -> bf16 conversion (weights / cond)
// ---------------------------------------------------------------------------
__global__ __launch_bounds__(256) void cvt_bf16(const float* __restrict__ in,
                                                unsigned short* __restrict__ out,
                                                int n) {
    int i = blockIdx.x * 256 + threadIdx.x;
    if (i < n) out[i] = f2bf(in[i]);
}

// ---------------------------------------------------------------------------
// GEMM: C(M,N) = A(M,K) @ W(N,K)^T (+bias), LDS double-buffered via async
// global->LDS copies.  Fused epilogues:
//   mode 0: C = acc (+bias)
//   mode 1: C = gelu_tanh(acc + bias)            (MLP hidden)
//   mode 2: C = resid + gate * (acc (+bias))     (adaLN residual)
// Block: 256 threads = 8 waves (2M x 4N), block tile 128x256, K-step 32.
// LDS: A stage 128x(32+8pad) halves, B stage 256x(32+8pad); 2 stages = 60 KB.
// Row stride 80B -> 16B-aligned chunks, conflict-free (20-bank stride).
// ---------------------------------------------------------------------------
#define LROW 40            // halves per LDS row (32 data + 8 pad)
#define A_ST (128 * LROW)  // halves per A stage
#define B_ST (256 * LROW)  // halves per B stage

__global__ __launch_bounds__(256) void gemm_bf16(
    const unsigned short* __restrict__ A, const unsigned short* __restrict__ Wm,
    const float* __restrict__ bias, const float* __restrict__ resid,
    const float* __restrict__ gate, int ldg,
    float* __restrict__ Cf, unsigned short* __restrict__ Cb,
    int M, int N, int K, int mode)
{
    __shared__ unsigned short smem[2 * A_ST + 2 * B_ST];   // 61440 B

    const unsigned int sbase = (unsigned int)(uintptr_t)smem;
    const int t    = threadIdx.x;
    const int wave = t >> 5;
    const int lane = t & 31;
    const int mw = (wave >> 2) * 64;            // wave row offset in block tile
    const int nw = (wave & 3) * 64;             // wave col offset in block tile
    const int m0b = blockIdx.y * 128;
    const int n0b = blockIdx.x * 256;

    // per-thread chunk coordinates for the cooperative stage copy
    // A tile: 512 chunks of 16B (2/thread); B tile: 1024 chunks (4/thread)
    v8f acc[4][4];
#pragma unroll
    for (int i = 0; i < 4; ++i)
#pragma unroll
        for (int j = 0; j < 4; ++j)
#pragma unroll
            for (int e = 0; e < 8; ++e) acc[i][j][e] = 0.0f;

    auto issue_stage = [&](int k0, int buf) {
        unsigned int aoff = sbase + (unsigned int)(buf * A_ST) * 2u;
        unsigned int boff = sbase + (unsigned int)(2 * A_ST + buf * B_ST) * 2u;
#pragma unroll
        for (int r = 0; r < 2; ++r) {
            int c = t + r * 256;                // 0..511
            int row = c >> 2, cc = c & 3;
            async_b128(aoff + (unsigned int)(row * (LROW * 2) + cc * 16),
                       A + (size_t)(m0b + row) * K + k0 + cc * 8);
        }
#pragma unroll
        for (int r = 0; r < 4; ++r) {
            int c = t + r * 256;                // 0..1023
            int row = c >> 2, cc = c & 3;
            async_b128(boff + (unsigned int)(row * (LROW * 2) + cc * 16),
                       Wm + (size_t)(n0b + row) * K + k0 + cc * 8);
        }
    };

    auto lds_frag = [&](const unsigned short* base, int row0) -> v16bf {
        int r  = row0 + (lane & 15);
        int kb = (lane >> 4) << 3;              // halves: 0 or 8
        const unsigned short* p = base + r * LROW + kb;
        FragU f;
        f.q[0] = *(const uint4*)(p);            // K {kb .. kb+7}
        f.q[1] = *(const uint4*)(p + 16);       // K {kb+16 .. kb+23}
        return f.v;
    };

    const int nStages = K >> 5;
    issue_stage(0, 0);

    for (int s = 0; s < nStages; ++s) {
        int buf = s & 1;
        if (s + 1 < nStages) {
            issue_stage((s + 1) << 5, buf ^ 1);
            asm volatile("s_wait_asynccnt 0x6" ::: "memory");  // stage s landed
        } else {
            asm volatile("s_wait_asynccnt 0x0" ::: "memory");
        }
        __syncthreads();                        // stage s visible block-wide

        const unsigned short* aBase = smem + buf * A_ST;
        const unsigned short* bBase = smem + 2 * A_ST + buf * B_ST;
        v16bf a[4], b[4];
#pragma unroll
        for (int i = 0; i < 4; ++i) a[i] = lds_frag(aBase, mw + 16 * i);
#pragma unroll
        for (int j = 0; j < 4; ++j) b[j] = lds_frag(bBase, nw + 16 * j);
#pragma unroll
        for (int i = 0; i < 4; ++i)
#pragma unroll
            for (int j = 0; j < 4; ++j)
                acc[i][j] = wmma_bf16(a[i], b[j], acc[i][j]);

        __syncthreads();                        // done reading buf before reuse
    }

    int cn = lane & 15, hi = lane >> 4;
    int m0 = m0b + mw, n0 = n0b + nw;
#pragma unroll
    for (int i = 0; i < 4; ++i) {
#pragma unroll
        for (int j = 0; j < 4; ++j) {
#pragma unroll
            for (int e = 0; e < 8; ++e) {
                int m = m0 + 16 * i + e + 8 * hi;
                int n = n0 + 16 * j + cn;
                float v = acc[i][j][e];
                if (bias) v += bias[n];
                if (mode == 1) {
                    float x = v;
                    float tt = tanhf(0.7978845608f * (x + 0.044715f * x * x * x));
                    v = 0.5f * x * (1.0f + tt);
                } else if (mode == 2) {
                    v = resid[(size_t)m * N + n] + gate[(size_t)m * ldg + n] * v;
                }
                size_t idx = (size_t)m * N + n;
                if (Cf) Cf[idx] = v;
                if (Cb) Cb[idx] = f2bf(v);
            }
        }
    }
}

// ---------------------------------------------------------------------------
// LayerNorm (+ optional adaLN modulate) -> bf16.  One block per row.
// ---------------------------------------------------------------------------
__global__ __launch_bounds__(256) void ln_mod(
    const float* __restrict__ x, const float* __restrict__ w,
    const float* __restrict__ shift, const float* __restrict__ scale, int ldg,
    unsigned short* __restrict__ out, int D)
{
    int row = blockIdx.x;
    const float* xr = x + (size_t)row * D;
    __shared__ float s1[256], s2[256];
    float a = 0.f, b = 0.f;
    for (int c = threadIdx.x; c < D; c += 256) { float v = xr[c]; a += v; b += v * v; }
    s1[threadIdx.x] = a; s2[threadIdx.x] = b;
    __syncthreads();
    for (int off = 128; off > 0; off >>= 1) {
        if (threadIdx.x < off) {
            s1[threadIdx.x] += s1[threadIdx.x + off];
            s2[threadIdx.x] += s2[threadIdx.x + off];
        }
        __syncthreads();
    }
    float mean = s1[0] / (float)D;
    float var  = s2[0] / (float)D - mean * mean;
    float rstd = rsqrtf(var + 1e-5f);
    unsigned short* orow = out + (size_t)row * D;
    const float* sh = shift ? shift + (size_t)row * ldg : nullptr;
    const float* sc = scale ? scale + (size_t)row * ldg : nullptr;
    for (int c = threadIdx.x; c < D; c += 256) {
        float v = (xr[c] - mean) * rstd * w[c];
        if (sc) v = v * (1.f + sc[c]) + sh[c];
        orow[c] = f2bf(v);
    }
}

// ---------------------------------------------------------------------------
// RoPE + pack kernels (fp32 GEMM output -> bf16 attention layouts).
//   q,k -> (B,H,N,HD) row-major;  v -> transposed (B,H,HD,L) so that PV
//   B-fragments are contiguous 16B loads.
// ---------------------------------------------------------------------------
__global__ __launch_bounds__(256) void pack_self(
    const float* __restrict__ qkv, const float* __restrict__ cosb,
    const float* __restrict__ sinb, unsigned short* __restrict__ qb,
    unsigned short* __restrict__ kb, unsigned short* __restrict__ vT)
{
    int idx = blockIdx.x * 256 + threadIdx.x;       // B*N*H*HD = 2^22
    int hd = idx & 63, h = (idx >> 6) & 15, n = (idx >> 10) & 1023, b = idx >> 20;
    size_t base = (size_t)(b * 1024 + n) * 3072 + h * 64 + hd;
    float c = cosb[n * 64 + hd], s = sinb[n * 64 + hd];
    int   hd2 = (hd < 32) ? hd + 32 : hd - 32;
    float sgn = (hd < 32) ? -1.f : 1.f;
    size_t base2 = base + (hd2 - hd);
    float qv = qkv[base]        * c + sgn * qkv[base2]        * s;
    float kv = qkv[base + 1024] * c + sgn * qkv[base2 + 1024] * s;
    float vv = qkv[base + 2048];
    size_t bh = (size_t)(b * 16 + h);
    qb[(bh * 1024 + n) * 64 + hd] = f2bf(qv);
    kb[(bh * 1024 + n) * 64 + hd] = f2bf(kv);
    vT[(bh * 64 + hd) * 1024 + n] = f2bf(vv);
}

__global__ __launch_bounds__(256) void pack_crossq(
    const float* __restrict__ qc, const float* __restrict__ cosb,
    const float* __restrict__ sinb, unsigned short* __restrict__ qb)
{
    int idx = blockIdx.x * 256 + threadIdx.x;       // B*N*H*HD
    int hd = idx & 63, h = (idx >> 6) & 15, n = (idx >> 10) & 1023, b = idx >> 20;
    size_t base = (size_t)(b * 1024 + n) * 1024 + h * 64 + hd;
    float c = cosb[n * 64 + hd], s = sinb[n * 64 + hd];
    int   hd2 = (hd < 32) ? hd + 32 : hd - 32;
    float sgn = (hd < 32) ? -1.f : 1.f;
    float qv = qc[base] * c + sgn * qc[base + (hd2 - hd)] * s;
    size_t bh = (size_t)(b * 16 + h);
    qb[(bh * 1024 + n) * 64 + hd] = f2bf(qv);
}

__global__ __launch_bounds__(256) void pack_crosskv(
    const float* __restrict__ kv, const float* __restrict__ cosb,
    const float* __restrict__ sinb, unsigned short* __restrict__ kb,
    unsigned short* __restrict__ vT)
{
    int idx = blockIdx.x * 256 + threadIdx.x;       // B*2N*H*HD = 2^23
    int hd = idx & 63, h = (idx >> 6) & 15, n2 = (idx >> 10) & 2047, b = idx >> 21;
    int pos = n2 & 1023;                            // both halves use positions 0..N-1
    size_t base = (size_t)(b * 2048 + n2) * 2048 + h * 64 + hd;
    float c = cosb[pos * 64 + hd], s = sinb[pos * 64 + hd];
    int   hd2 = (hd < 32) ? hd + 32 : hd - 32;
    float sgn = (hd < 32) ? -1.f : 1.f;
    float kvv = kv[base] * c + sgn * kv[base + (hd2 - hd)] * s;
    float vv  = kv[base + 1024];
    size_t bh = (size_t)(b * 16 + h);
    kb[(bh * 2048 + n2) * 64 + hd]  = f2bf(kvv);
    vT[(bh * 64 + hd) * 2048 + n2] = f2bf(vv);
}

// ---------------------------------------------------------------------------
// Flash attention. Q:(B,H,N,64) bf16, Kt:(B,H,Lk,64) bf16, Vt:(B,H,64,Lk) bf16,
// mask:(B,N,Lk) f32 additive.  Out:(B,N,H*64) bf16 (row = b*N+n, ready as GEMM A).
// Block = 128 threads (4 waves); each wave owns 16 query rows; keys in blocks
// of 32 (2 score tiles of 16 cols -> 8 WMMAs / 32 keys incl. PV).
// ---------------------------------------------------------------------------
__global__ __launch_bounds__(128) void attn_fwd(
    const unsigned short* __restrict__ Q, const unsigned short* __restrict__ Kt,
    const unsigned short* __restrict__ Vt, const float* __restrict__ mask,
    float scl, int Lk, unsigned short* __restrict__ Out)
{
    const int N = 1024, HD = 64;
    int wave = threadIdx.x >> 5, lane = threadIdx.x & 31;
    int cn = lane & 15, hi = lane >> 4;
    int bh = blockIdx.y, b = bh >> 4, h = bh & 15;
    int q0 = blockIdx.x * 64 + wave * 16;

    const unsigned short* qp = Q  + (size_t)bh * N  * HD;
    const unsigned short* kp = Kt + (size_t)bh * Lk * HD;
    const unsigned short* vp = Vt + (size_t)bh * HD * Lk;
    const float*          mp = mask + (size_t)b * N * Lk;

    v16bf aQ0 = load_frag(qp, q0, 0,  HD);
    v16bf aQ1 = load_frag(qp, q0, 32, HD);

    float rmax[8], rsum[8];
    v8f accO[4];
#pragma unroll
    for (int e = 0; e < 8; ++e) { rmax[e] = -3.0e38f; rsum[e] = 0.f; }
#pragma unroll
    for (int t = 0; t < 4; ++t)
#pragma unroll
        for (int e = 0; e < 8; ++e) accO[t][e] = 0.f;

    __shared__ unsigned short Pl[4][16][40];   // per-wave P staging, padded

    for (int j0 = 0; j0 < Lk; j0 += 32) {
        v16bf bA0 = load_frag(kp, j0,      0,  HD);
        v16bf bA1 = load_frag(kp, j0,      32, HD);
        v16bf bB0 = load_frag(kp, j0 + 16, 0,  HD);
        v16bf bB1 = load_frag(kp, j0 + 16, 32, HD);
        v8f z0, z1;
#pragma unroll
        for (int e = 0; e < 8; ++e) { z0[e] = 0.f; z1[e] = 0.f; }
        z0 = wmma_bf16(aQ0, bA0, z0);  z0 = wmma_bf16(aQ1, bA1, z0);
        z1 = wmma_bf16(aQ0, bB0, z1);  z1 = wmma_bf16(aQ1, bB1, z1);

#pragma unroll
        for (int e = 0; e < 8; ++e) {
            int   qr = q0 + e + 8 * hi;
            float s0 = z0[e] * scl + mp[(size_t)qr * Lk + j0 + cn];
            float s1 = z1[e] * scl + mp[(size_t)qr * Lk + j0 + 16 + cn];
            float t  = fmaxf(s0, s1);
#pragma unroll
            for (int d = 1; d < 16; d <<= 1) t = fmaxf(t, __shfl_xor(t, d, 16));
            float nm = fmaxf(rmax[e], t);
            float al = __expf(rmax[e] - nm);
            rmax[e]  = nm;
            float e0 = __expf(s0 - nm), e1 = __expf(s1 - nm);
            float ps = e0 + e1;
#pragma unroll
            for (int d = 1; d < 16; d <<= 1) ps += __shfl_xor(ps, d, 16);
            rsum[e] = rsum[e] * al + ps;
#pragma unroll
            for (int t4 = 0; t4 < 4; ++t4) accO[t4][e] *= al;
            int m = e + 8 * hi;
            Pl[wave][m][cn]      = f2bf(e0);
            Pl[wave][m][16 + cn] = f2bf(e1);
        }
        __syncthreads();     // cross-lane LDS visibility (uniform across block)

        // Re-read P as a 16x32 A-fragment.
        FragU pu;
        int r = lane & 15, kb = hi * 8;
#pragma unroll
        for (int i = 0; i < 8; ++i) pu.u[i]     = Pl[wave][r][kb + i];
#pragma unroll
        for (int i = 0; i < 8; ++i) pu.u[8 + i] = Pl[wave][r][16 + kb + i];

#pragma unroll
        for (int t = 0; t < 4; ++t) {
            v16bf bV = load_frag(vp, t * 16, j0, Lk);   // rows = hd, K = key idx
            accO[t]  = wmma_bf16(pu.v, bV, accO[t]);
        }
    }

#pragma unroll
    for (int e = 0; e < 8; ++e) { float s = rsum[e]; rsum[e] = (s > 0.f) ? 1.f / s : 0.f; }
#pragma unroll
    for (int t = 0; t < 4; ++t)
#pragma unroll
        for (int e = 0; e < 8; ++e) {
            int qr = q0 + e + 8 * hi;
            int hd = t * 16 + cn;
            Out[((size_t)b * N + qr) * 1024 + h * 64 + hd] = f2bf(accO[t][e] * rsum[e]);
        }
}

// ---------------------------------------------------------------------------
// Host orchestration.
// ---------------------------------------------------------------------------
extern "C" void kernel_launch(void* const* d_in, const int* in_sizes, int n_in,
                              void* d_out, int out_size, void* d_ws, size_t ws_size,
                              hipStream_t stream)
{
    (void)in_sizes; (void)n_in; (void)out_size; (void)ws_size;

    const float* q_x        = (const float*)d_in[0];
    const float* h_content  = (const float*)d_in[1];
    const float* h_obs      = (const float*)d_in[2];
    const float* t_cond     = (const float*)d_in[3];
    const float* cosb       = (const float*)d_in[4];
    const float* sinb       = (const float*)d_in[5];
    const float* M_QQ       = (const float*)d_in[6];
    const float* M_hyb      = (const float*)d_in[7];
    const float* w_ln_self  = (const float*)d_in[8];
    const float* w_qkv      = (const float*)d_in[9];
    const float* w_self_out = (const float*)d_in[10];
    const float* w_ln_cross = (const float*)d_in[11];
    const float* w_ln_mem   = (const float*)d_in[12];
    const float* w_qproj    = (const float*)d_in[13];
    const float* w_kvproj   = (const float*)d_in[14];
    const float* w_cross_out= (const float*)d_in[15];
    const float* w_ln_mlp   = (const float*)d_in[16];
    const float* w_mlp1     = (const float*)d_in[17];
    const float* b_mlp1     = (const float*)d_in[18];
    const float* w_mlp2     = (const float*)d_in[19];
    const float* b_mlp2     = (const float*)d_in[20];
    const float* w_ada      = (const float*)d_in[21];
    const float* b_ada      = (const float*)d_in[22];

    const int Bt = 4, N = 1024, D = 1024, COND = 256;
    const int M  = Bt * N;        // 4096 token rows
    const int M2 = 2 * M;         // 8192 memory rows
    const int A9 = 9 * D;         // 9216 used adaLN features

    char*  ws  = (char*)d_ws;
    size_t off = 0;
    auto alloc = [&](size_t bytes) -> void* {
        void* p = ws + off;
        off = (off + bytes + 255) & ~(size_t)255;
        return p;
    };
    auto bfbuf = [&](size_t n) { return (unsigned short*)alloc(n * 2); };
    auto f32buf= [&](size_t n) { return (float*)alloc(n * 4); };

    // bf16 weights / inputs
    unsigned short* tcond_bf  = bfbuf((size_t)M * COND);
    unsigned short* wada_bf   = bfbuf((size_t)A9 * COND);
    unsigned short* wqkv_bf   = bfbuf((size_t)3 * D * D);
    unsigned short* wself_bf  = bfbuf((size_t)D * D);
    unsigned short* wqproj_bf = bfbuf((size_t)D * D);
    unsigned short* wkv_bf    = bfbuf((size_t)2 * D * D);
    unsigned short* wcross_bf = bfbuf((size_t)D * D);
    unsigned short* wmlp1_bf  = bfbuf((size_t)4 * D * D);
    unsigned short* wmlp2_bf  = bfbuf((size_t)4 * D * D);
    // bf16 activations
    unsigned short* xn_bf     = bfbuf((size_t)M * D);
    unsigned short* qb        = bfbuf((size_t)M * D);
    unsigned short* kbuf      = bfbuf((size_t)M * D);
    unsigned short* vT        = bfbuf((size_t)M * D);
    unsigned short* attn_bf   = bfbuf((size_t)M * D);
    unsigned short* mem_bf    = bfbuf((size_t)M2 * D);
    unsigned short* kc_bf     = bfbuf((size_t)M2 * D);
    unsigned short* vcT       = bfbuf((size_t)M2 * D);
    unsigned short* h_bf      = bfbuf((size_t)M * 4 * D);
    // f32 intermediates
    float* ada     = f32buf((size_t)M * A9);
    float* qkv_f32 = f32buf((size_t)M * 3 * D);
    float* qc_f32  = f32buf((size_t)M * D);
    float* kv_f32  = f32buf((size_t)M2 * 2 * D);
    float* qx_a    = f32buf((size_t)M * D);
    float* qx_b    = f32buf((size_t)M * D);

    auto cvt = [&](const float* s, unsigned short* dst, int n) {
        cvt_bf16<<<(n + 255) / 256, 256, 0, stream>>>(s, dst, n);
    };
    auto gemm = [&](const unsigned short* A, const unsigned short* W,
                    const float* bias, const float* resid, const float* gate,
                    int ldg, float* Cf, unsigned short* Cb,
                    int Mi, int Ni, int Ki, int mode) {
        dim3 g(Ni / 256, Mi / 128);
        gemm_bf16<<<g, 256, 0, stream>>>(A, W, bias, resid, gate, ldg,
                                         Cf, Cb, Mi, Ni, Ki, mode);
    };

    // 0) fp32 -> bf16 for everything feeding matrix cores
    cvt(t_cond,     tcond_bf,  M * COND);
    cvt(w_ada,      wada_bf,   A9 * COND);     // only the 9 used chunks
    cvt(w_qkv,      wqkv_bf,   3 * D * D);
    cvt(w_self_out, wself_bf,  D * D);
    cvt(w_qproj,    wqproj_bf, D * D);
    cvt(w_kvproj,   wkv_bf,    2 * D * D);
    cvt(w_cross_out,wcross_bf, D * D);
    cvt(w_mlp1,     wmlp1_bf,  4 * D * D);
    cvt(w_mlp2,     wmlp2_bf,  4 * D * D);

    // 1) ada = t_cond @ w_ada^T + b_ada   (first 9 chunks)
    gemm(tcond_bf, wada_bf, b_ada, nullptr, nullptr, 0, ada, nullptr,
         M, A9, COND, 0);

    // 2) self-attention branch
    ln_mod<<<M, 256, 0, stream>>>(q_x, w_ln_self, ada + 0 * D, ada + 1 * D, A9,
                                  xn_bf, D);
    gemm(xn_bf, wqkv_bf, nullptr, nullptr, nullptr, 0, qkv_f32, nullptr,
         M, 3 * D, D, 0);
    pack_self<<<(M * D) / 256, 256, 0, stream>>>(qkv_f32, cosb, sinb, qb, kbuf, vT);
    attn_fwd<<<dim3(N / 64, Bt * 16), 128, 0, stream>>>(qb, kbuf, vT, M_QQ,
                                                        0.125f, N, attn_bf);
    gemm(attn_bf, wself_bf, nullptr, q_x, ada + 2 * D, A9, qx_a, nullptr,
         M, D, D, 2);

    // 3) hybrid cross-attention branch
    ln_mod<<<M, 256, 0, stream>>>(qx_a, w_ln_cross, ada + 3 * D, ada + 4 * D, A9,
                                  xn_bf, D);
    gemm(xn_bf, wqproj_bf, nullptr, nullptr, nullptr, 0, qc_f32, nullptr,
         M, D, D, 0);
    ln_mod<<<M, 256, 0, stream>>>(h_content, w_ln_mem, nullptr, nullptr, 0,
                                  mem_bf, D);
    ln_mod<<<M, 256, 0, stream>>>(h_obs, w_ln_mem, nullptr, nullptr, 0,
                                  mem_bf + (size_t)M * D, D);
    gemm(mem_bf, wkv_bf, nullptr, nullptr, nullptr, 0, kv_f32, nullptr,
         M2, 2 * D, D, 0);
    pack_crossq<<<(M * D) / 256, 256, 0, stream>>>(qc_f32, cosb, sinb, qb);
    pack_crosskv<<<(M2 * D) / 256, 256, 0, stream>>>(kv_f32, cosb, sinb, kc_bf, vcT);
    attn_fwd<<<dim3(N / 64, Bt * 16), 128, 0, stream>>>(qb, kc_bf, vcT, M_hyb,
                                                        0.125f, 2 * N, attn_bf);
    gemm(attn_bf, wcross_bf, nullptr, qx_a, ada + 5 * D, A9, qx_b, nullptr,
         M, D, D, 2);

    // 4) MLP branch
    ln_mod<<<M, 256, 0, stream>>>(qx_b, w_ln_mlp, ada + 6 * D, ada + 7 * D, A9,
                                  xn_bf, D);
    gemm(xn_bf, wmlp1_bf, b_mlp1, nullptr, nullptr, 0, nullptr, h_bf,
         M, 4 * D, D, 1);
    gemm(h_bf, wmlp2_bf, b_mlp2, qx_b, ada + 8 * D, A9, (float*)d_out, nullptr,
         M, D, 4 * D, 2);
}
